// Attention_6743098655262
// MI455X (gfx1250) — matrix operand order repaired and checked by reading, not compiled, not run on hardware
//
#include <hip/hip_runtime.h>

// ---------------------------------------------------------------------------
// Problem constants (match reference)
// ---------------------------------------------------------------------------
constexpr int DIM     = 2048;
constexpr int N_HEADS = 32;
constexpr int N_KV    = 8;
constexpr int HD      = 64;          // head dim
constexpr int BATCH   = 2;
constexpr int SEQ     = 2048;
constexpr int QKV_N   = (N_HEADS + 2 * N_KV) * HD;   // 3072
constexpr int MTOT    = BATCH * SEQ;                 // 4096

typedef __attribute__((ext_vector_type(16))) __bf16 bf16x16;
typedef __attribute__((ext_vector_type(8)))  float  floatx8;
typedef __attribute__((ext_vector_type(4)))  unsigned int u32x4;
typedef __attribute__((ext_vector_type(8)))  int          i32x8;
typedef __attribute__((ext_vector_type(4)))  int          i32x4;

union FragBF16 {
    bf16x16        v;
    unsigned int   u[8];
    unsigned short s[16];
};

__device__ __forceinline__ unsigned short f2bf(float f) {
    unsigned int u = __float_as_uint(f);
    u += 0x7fffu + ((u >> 16) & 1u);   // round-to-nearest-even
    return (unsigned short)(u >> 16);
}

__device__ __forceinline__ floatx8 zero8() {
    floatx8 z;
#pragma unroll
    for (int e = 0; e < 8; ++e) z[e] = 0.0f;
    return z;
}

__device__ __forceinline__ float rowmax16(float v) {
    v = fmaxf(v, __shfl_xor(v, 1, 32));
    v = fmaxf(v, __shfl_xor(v, 2, 32));
    v = fmaxf(v, __shfl_xor(v, 4, 32));
    v = fmaxf(v, __shfl_xor(v, 8, 32));
    return v;
}
__device__ __forceinline__ float rowsum16(float v) {
    v += __shfl_xor(v, 1, 32);
    v += __shfl_xor(v, 2, 32);
    v += __shfl_xor(v, 4, 32);
    v += __shfl_xor(v, 8, 32);
    return v;
}

// ---------------------------------------------------------------------------
// TDM: 2D tensor_load_to_lds of a [ty rows x tx elems] bf16 tile starting at
// gptr (row stride = ld elements, tensor = nrows x ld) into LDS byte offset.
// D# bitfields per CDNA5 ISA ch.8.
// ---------------------------------------------------------------------------
__device__ __forceinline__ void tdm_load_tile(const unsigned short* gptr,
                                              unsigned int lds_off,
                                              int ld, int nrows,
                                              int tx, int ty) {
    unsigned long long ga = (unsigned long long)(uintptr_t)gptr;  // byte addr
    u32x4 g0;
    g0[0] = 1u;                                        // count=1, user mode
    g0[1] = lds_off;                                   // lds_addr (bytes)
    g0[2] = (unsigned int)(ga & 0xffffffffu);          // global_addr[31:0]
    g0[3] = (unsigned int)((ga >> 32) & 0x01ffffffu)   // global_addr[56:32]
          | (2u << 30);                                // type = 2 ("image")
    i32x8 g1;
    g1[0] = (int)(1u << 16);                           // data_size=1 -> 2 bytes
    g1[1] = (int)(((unsigned)ld & 0xffffu) << 16);     // tensor_dim0[15:0]
    g1[2] = (int)((((unsigned)nrows & 0xffffu) << 16)  // tensor_dim1[15:0]
          | (((unsigned)ld >> 16) & 0xffffu));         // tensor_dim0[31:16]
    g1[3] = (int)((((unsigned)tx) << 16)               // tile_dim0
          | (((unsigned)nrows >> 16) & 0xffffu));      // tensor_dim1[31:16]
    g1[4] = ty;                                        // tile_dim1
    g1[5] = ld;                                        // tensor_dim0_stride lo32
    g1[6] = 0;
    g1[7] = 0;
    i32x4 g2, g3;
#pragma unroll
    for (int i = 0; i < 4; ++i) { g2[i] = 0; g3[i] = 0; }
#if defined(__clang_major__) && (__clang_major__ >= 23)
    i32x8 g4;
#pragma unroll
    for (int i = 0; i < 8; ++i) g4[i] = 0;
    __builtin_amdgcn_tensor_load_to_lds(g0, g1, g2, g3, g4, 0);
#else
    __builtin_amdgcn_tensor_load_to_lds(g0, g1, g2, g3, 0);
#endif
}

// ---------------------------------------------------------------------------
// Elementwise f32 -> bf16 conversion
// ---------------------------------------------------------------------------
__global__ void cvt_bf16_kernel(const float* __restrict__ src,
                                unsigned short* __restrict__ dst, int n) {
    int i = blockIdx.x * 256 + threadIdx.x;
    if (i < n) dst[i] = f2bf(src[i]);
}

// ---------------------------------------------------------------------------
// Tiled bf16 GEMM: C[M,N](f32) = A[M,K](bf16 rm) * Bt[N,K](bf16 rm)^T
// Block tile 128x128, 8 waves as 4(M)x2(N); wave tile 32x64 -> 8 WMMA/K-step.
// Both operand tiles are [128 rows x 32 k] staged by TDM (tensor_load_to_lds),
// double-buffered: the hot loop is purely TDM -> ds_load_b128 -> WMMA.
// ---------------------------------------------------------------------------
__global__ __launch_bounds__(256)
void gemm_bf16_kernel(const unsigned short* __restrict__ A,
                      const unsigned short* __restrict__ Bt,
                      float* __restrict__ C, int M, int N, int K) {
    __shared__ unsigned short ldsA[2][128][32];    // [buf][m][k]
    __shared__ unsigned short ldsB[2][128][32];    // [buf][n][k]

    const int tid  = threadIdx.x;
    const int lane = tid & 31;
    const int w    = tid >> 5;
    const int mrow = lane & 15;
    const int hh   = lane >> 4;
    const int wm   = w & 3;          // 0..3 -> M sub-block
    const int wn   = w >> 2;         // 0..1 -> N sub-block
    const int m_blk = blockIdx.y * 128;
    const int n_blk = blockIdx.x * 128;

    const unsigned int aoff[2] = {
        (unsigned int)(uintptr_t)(&ldsA[0][0][0]),
        (unsigned int)(uintptr_t)(&ldsA[1][0][0])
    };
    const unsigned int boff[2] = {
        (unsigned int)(uintptr_t)(&ldsB[0][0][0]),
        (unsigned int)(uintptr_t)(&ldsB[1][0][0])
    };

    floatx8 acc[2][4];
#pragma unroll
    for (int m = 0; m < 2; ++m)
#pragma unroll
        for (int t = 0; t < 4; ++t) acc[m][t] = zero8();

    // ---- prologue: stage K-tile 0 into buffer 0 via TDM
    if (w == 0) tdm_load_tile(A  + (size_t)m_blk * K, aoff[0], K, M, 32, 128);
    if (w == 1) tdm_load_tile(Bt + (size_t)n_blk * K, boff[0], K, N, 32, 128);
    __builtin_amdgcn_s_wait_tensorcnt(0);
    __syncthreads();

    for (int k0 = 0; k0 < K; k0 += 32) {
        const int cur  = (k0 >> 5) & 1;
        const int nxt  = cur ^ 1;
        const bool more = (k0 + 32) < K;

        // ---- prefetch next K-tile straight into the other LDS buffer
        if (more) {
            if (w == 0)
                tdm_load_tile(A  + (size_t)m_blk * K + (k0 + 32), aoff[nxt], K, M, 32, 128);
            if (w == 1)
                tdm_load_tile(Bt + (size_t)n_blk * K + (k0 + 32), boff[nxt], K, N, 32, 128);
        }

        // ---- A fragments (two 16x32 sub-tiles per wave)
        FragBF16 af[2];
#pragma unroll
        for (int m = 0; m < 2; ++m) {
            const unsigned short* arow = &ldsA[cur][wm * 32 + m * 16 + mrow][0];
#pragma unroll
            for (int j = 0; j < 8; ++j) {
                int kk = ((j >= 4) ? 16 : 0) + hh * 8 + (j & 3) * 2;
                af[m].u[j] = *(const unsigned int*)(arow + kk);
            }
        }
        // ---- 4 B fragments preloaded, then an 8-WMMA burst
        FragBF16 bf4[4];
#pragma unroll
        for (int t = 0; t < 4; ++t) {
            const unsigned short* brow = &ldsB[cur][wn * 64 + t * 16 + mrow][0];
#pragma unroll
            for (int j = 0; j < 8; ++j)
                bf4[t].u[j] = *(const unsigned int*)(brow + ((j >= 4) ? 16 : 0) + hh * 8 + (j & 3) * 2);
        }
#pragma unroll
        for (int t = 0; t < 4; ++t)
#pragma unroll
            for (int m = 0; m < 2; ++m)
                acc[m][t] = __builtin_amdgcn_wmma_f32_16x16x32_bf16(
                    false, af[m].v, false, bf4[t].v, (short)0, acc[m][t], false, false);

        __builtin_amdgcn_s_wait_tensorcnt(0);
        __syncthreads();
    }

    // ---- writeback (C layout: vgpr v -> row v + 8*hh, lane%16 -> col)
#pragma unroll
    for (int m = 0; m < 2; ++m)
#pragma unroll
        for (int t = 0; t < 4; ++t)
#pragma unroll
            for (int v = 0; v < 8; ++v) {
                int row = m_blk + wm * 32 + m * 16 + v + 8 * hh;
                int col = n_blk + wn * 64 + t * 16 + mrow;
                C[(size_t)row * N + col] = acc[m][t][v];
            }
}

// ---------------------------------------------------------------------------
// RoPE on q/k portions of the fp32 qkv buffer, emit bf16 head-major layouts.
// q -> [B, H, S, hd], k -> [B, G, S, hd]
// ---------------------------------------------------------------------------
__global__ void rope_qk_kernel(const float* __restrict__ qkv,
                               const float* __restrict__ fcos,
                               const float* __restrict__ fsin,
                               unsigned short* __restrict__ qb,
                               unsigned short* __restrict__ kb) {
    const int PAIRS = HD / 2;                 // 32
    const int HEADS = N_HEADS + N_KV;         // 40
    int idx = blockIdx.x * 256 + threadIdx.x;
    int total = BATCH * SEQ * HEADS * PAIRS;
    if (idx >= total) return;

    int t = idx % PAIRS;       int r = idx / PAIRS;
    int head = r % HEADS;      r /= HEADS;
    int s = r % SEQ;           int b = r / SEQ;
    int d0 = 2 * t;

    int colbase = (head < N_HEADS) ? head * HD : DIM + (head - N_HEADS) * HD;
    const float* xr = qkv + ((size_t)(b * SEQ + s)) * QKV_N + colbase;
    float x0 = xr[d0], x1 = xr[d0 + 1];
    float c  = fcos[s * HD + d0];
    float sn = fsin[s * HD + d0];
    float o0 = x0 * c - x1 * sn;
    float o1 = x1 * c + x0 * sn;

    if (head < N_HEADS) {
        unsigned short* orow = qb + (((size_t)(b * N_HEADS + head)) * SEQ + s) * HD;
        orow[d0] = f2bf(o0);  orow[d0 + 1] = f2bf(o1);
    } else {
        int g = head - N_HEADS;
        unsigned short* orow = kb + (((size_t)(b * N_KV + g)) * SEQ + s) * HD;
        orow[d0] = f2bf(o0);  orow[d0 + 1] = f2bf(o1);
    }
}

// v -> transposed bf16 [B, G, hd, S] so P@V B-fragments are contiguous pairs
__global__ void v_transpose_kernel(const float* __restrict__ qkv,
                                   unsigned short* __restrict__ vT) {
    int idx = blockIdx.x * 256 + threadIdx.x;
    int total = BATCH * N_KV * SEQ * HD;
    if (idx >= total) return;
    int d = idx % HD;  int r = idx / HD;
    int s = r % SEQ;   r /= SEQ;
    int g = r % N_KV;  int b = r / N_KV;
    float val = qkv[((size_t)(b * SEQ + s)) * QKV_N + DIM + N_KV * HD + g * HD + d];
    vT[(((size_t)(b * N_KV + g)) * HD + d) * SEQ + s] = f2bf(val);
}

// ---------------------------------------------------------------------------
// Flash-style attention. One block = one (b,h); its 8 waves cover 8
// consecutive 16-row q-blocks so they share K/V. K and V^T tiles (64 keys)
// are staged into double-buffered LDS by TDM; waves run
// QK^T WMMA -> online softmax -> P relayout (per-wave LDS) -> PV WMMA.
// ---------------------------------------------------------------------------
__global__ __launch_bounds__(256)
void attn_kernel(const unsigned short* __restrict__ qb,
                 const unsigned short* __restrict__ kb,
                 const unsigned short* __restrict__ vT,
                 unsigned short* __restrict__ ob) {
    __shared__ unsigned short ldsK[2][64][64];    // [buf][key][d]
    __shared__ unsigned short ldsVt[2][64][64];   // [buf][d][key]
    __shared__ unsigned short ldsP[8][16][64];    // per-wave P scratch

    const int tid  = threadIdx.x;
    const int lane = tid & 31;
    const int w    = tid >> 5;
    const int mrow = lane & 15;
    const int hh   = lane >> 4;

    const int QCH = SEQ / 128;                    // 16 q-chunks per (b,h)
    int qchunk = blockIdx.x % QCH;
    int bh     = blockIdx.x / QCH;
    int h      = bh % N_HEADS;
    int b      = bh / N_HEADS;
    int g      = h / (N_HEADS / N_KV);
    int q0     = qchunk * 128 + w * 16;           // this wave's 16 query rows

    const unsigned int koff[2] = {
        (unsigned int)(uintptr_t)(&ldsK[0][0][0]),
        (unsigned int)(uintptr_t)(&ldsK[1][0][0])
    };
    const unsigned int voff[2] = {
        (unsigned int)(uintptr_t)(&ldsVt[0][0][0]),
        (unsigned int)(uintptr_t)(&ldsVt[1][0][0])
    };

    // --- Q fragments (kept in registers for the whole key loop)
    FragBF16 qf[2];
    const unsigned short* qrow =
        qb + (((size_t)(b * N_HEADS + h)) * SEQ + q0 + mrow) * HD;
#pragma unroll
    for (int ks = 0; ks < 2; ++ks)
#pragma unroll
        for (int j = 0; j < 8; ++j) {
            int kk = ks * 32 + ((j >= 4) ? 16 : 0) + hh * 8 + (j & 3) * 2;
            qf[ks].u[j] = *(const unsigned int*)(qrow + kk);
        }

    const unsigned short* kbase = kb + ((size_t)(b * N_KV + g)) * SEQ * HD;
    const unsigned short* vbase = vT + ((size_t)(b * N_KV + g)) * HD * SEQ;

    float Mrow[8], Lrow[8];
#pragma unroll
    for (int v = 0; v < 8; ++v) { Mrow[v] = -3.0e38f; Lrow[v] = 0.0f; }
    floatx8 Oacc[4];
#pragma unroll
    for (int t = 0; t < 4; ++t) Oacc[t] = zero8();

    const float scale = 0.125f;                   // 1/sqrt(64)

    // ---- prologue: stage first 64-key K and V^T tiles
    if (w == 0) tdm_load_tile(kbase, koff[0], HD, SEQ, 64, 64);
    if (w == 1) tdm_load_tile(vbase, voff[0], SEQ, HD, 64, 64);
    __builtin_amdgcn_s_wait_tensorcnt(0);
    __syncthreads();

    for (int s0 = 0; s0 < SEQ; s0 += 64) {
        const int cur  = (s0 >> 6) & 1;
        const int nxt  = cur ^ 1;
        const bool more = (s0 + 64) < SEQ;

        if (more) {
            if (w == 0)
                tdm_load_tile(kbase + (size_t)(s0 + 64) * HD, koff[nxt], HD, SEQ, 64, 64);
            if (w == 1)
                tdm_load_tile(vbase + (s0 + 64), voff[nxt], SEQ, HD, 64, 64);
        }

        // ----- scores: 4 tiles of 16x16, hd reduced via 2 WMMA k-steps
        floatx8 Sacc[4];
#pragma unroll
        for (int t = 0; t < 4; ++t) {
            const unsigned short* krow = &ldsK[cur][t * 16 + mrow][0];
            FragBF16 kf0, kf1;
#pragma unroll
            for (int j = 0; j < 8; ++j) {
                kf0.u[j] = *(const unsigned int*)(krow + ((j >= 4) ? 16 : 0) + hh * 8 + (j & 3) * 2);
                kf1.u[j] = *(const unsigned int*)(krow + 32 + ((j >= 4) ? 16 : 0) + hh * 8 + (j & 3) * 2);
            }
            floatx8 z = zero8();
            z = __builtin_amdgcn_wmma_f32_16x16x32_bf16(
                false, qf[0].v, false, kf0.v, (short)0, z, false, false);
            Sacc[t] = __builtin_amdgcn_wmma_f32_16x16x32_bf16(
                false, qf[1].v, false, kf1.v, (short)0, z, false, false);
        }

        // ----- online softmax per row (row = vgpr index + 8*hh)
#pragma unroll
        for (int v = 0; v < 8; ++v) {
            float mx = Mrow[v];
#pragma unroll
            for (int t = 0; t < 4; ++t) {
                float x = Sacc[t][v] * scale;
                Sacc[t][v] = x;
                mx = fmaxf(mx, rowmax16(x));
            }
            float fac = __expf(Mrow[v] - mx);
            float sum = 0.0f;
#pragma unroll
            for (int t = 0; t < 4; ++t) {
                float p = __expf(Sacc[t][v] - mx);
                Sacc[t][v] = p;
                sum += p;
            }
            sum = rowsum16(sum);
            Lrow[v] = Lrow[v] * fac + sum;
            Mrow[v] = mx;
#pragma unroll
            for (int t = 0; t < 4; ++t) Oacc[t][v] *= fac;
        }

        // ----- relayout P (C layout) -> A-fragment layout via per-wave LDS
#pragma unroll
        for (int t = 0; t < 4; ++t)
#pragma unroll
            for (int v = 0; v < 8; ++v)
                ldsP[w][v + 8 * hh][t * 16 + mrow] = f2bf(Sacc[t][v]);

        FragBF16 pf[2];
#pragma unroll
        for (int ks = 0; ks < 2; ++ks)
#pragma unroll
            for (int j = 0; j < 8; ++j) {
                int kk = ks * 32 + ((j >= 4) ? 16 : 0) + hh * 8 + (j & 3) * 2;
                pf[ks].u[j] = *(const unsigned int*)&ldsP[w][mrow][kk];
            }

        // ----- O += P @ V   (V^T rows contiguous along keys)
#pragma unroll
        for (int t = 0; t < 4; ++t) {
            const unsigned short* vrow = &ldsVt[cur][t * 16 + mrow][0];
            FragBF16 vf0, vf1;
#pragma unroll
            for (int j = 0; j < 8; ++j) {
                vf0.u[j] = *(const unsigned int*)(vrow + ((j >= 4) ? 16 : 0) + hh * 8 + (j & 3) * 2);
                vf1.u[j] = *(const unsigned int*)(vrow + 32 + ((j >= 4) ? 16 : 0) + hh * 8 + (j & 3) * 2);
            }
            Oacc[t] = __builtin_amdgcn_wmma_f32_16x16x32_bf16(
                false, pf[0].v, false, vf0.v, (short)0, Oacc[t], false, false);
            Oacc[t] = __builtin_amdgcn_wmma_f32_16x16x32_bf16(
                false, pf[1].v, false, vf1.v, (short)0, Oacc[t], false, false);
        }

        __builtin_amdgcn_s_wait_tensorcnt(0);
        __syncthreads();
    }

    // ----- normalize and store bf16 o in [B, S, H*hd] (= GEMM2 A layout)
#pragma unroll
    for (int t = 0; t < 4; ++t)
#pragma unroll
        for (int v = 0; v < 8; ++v) {
            float val = Oacc[t][v] / Lrow[v];
            int srow = q0 + v + 8 * hh;
            ob[((size_t)(b * SEQ + srow)) * DIM + h * HD + t * 16 + mrow] = f2bf(val);
        }
}

// ---------------------------------------------------------------------------
// Host-side launcher
// ---------------------------------------------------------------------------
extern "C" void kernel_launch(void* const* d_in, const int* in_sizes, int n_in,
                              void* d_out, int out_size, void* d_ws, size_t ws_size,
                              hipStream_t stream) {
    (void)in_sizes; (void)n_in; (void)out_size; (void)ws_size;

    const float* x    = (const float*)d_in[0];
    const float* fcos = (const float*)d_in[1];
    const float* fsin = (const float*)d_in[2];
    const float* wqkv = (const float*)d_in[3];
    const float* wo   = (const float*)d_in[4];
    float* out = (float*)d_out;

    char* ws = (char*)d_ws;
    size_t off = 0;
    auto alloc = [&](size_t bytes) -> char* {
        char* p = ws + off;
        off += (bytes + 255) & ~(size_t)255;
        return p;
    };
    unsigned short* xb    = (unsigned short*)alloc((size_t)MTOT * DIM * 2);
    unsigned short* wqkvb = (unsigned short*)alloc((size_t)QKV_N * DIM * 2);  // [N][K]
    unsigned short* wob   = (unsigned short*)alloc((size_t)DIM * DIM * 2);    // [N][K]
    float*          qkvf  = (float*)alloc((size_t)MTOT * QKV_N * 4);
    unsigned short* qb    = (unsigned short*)alloc((size_t)BATCH * N_HEADS * SEQ * HD * 2);
    unsigned short* kb    = (unsigned short*)alloc((size_t)BATCH * N_KV * SEQ * HD * 2);
    unsigned short* vTb   = (unsigned short*)alloc((size_t)BATCH * N_KV * HD * SEQ * 2);
    unsigned short* obuf  = (unsigned short*)alloc((size_t)MTOT * DIM * 2);

    int n1 = MTOT * DIM;
    cvt_bf16_kernel<<<(n1 + 255) / 256, 256, 0, stream>>>(x, xb, n1);
    int n2 = QKV_N * DIM;
    cvt_bf16_kernel<<<(n2 + 255) / 256, 256, 0, stream>>>(wqkv, wqkvb, n2);
    int n3 = DIM * DIM;
    cvt_bf16_kernel<<<(n3 + 255) / 256, 256, 0, stream>>>(wo, wob, n3);

    // QKV projection: [4096,2048] x [2048,3072] -> f32 qkv
    gemm_bf16_kernel<<<dim3(QKV_N / 128, MTOT / 128), 256, 0, stream>>>(
        xb, wqkvb, qkvf, MTOT, QKV_N, DIM);

    int nr = BATCH * SEQ * (N_HEADS + N_KV) * (HD / 2);
    rope_qk_kernel<<<(nr + 255) / 256, 256, 0, stream>>>(qkvf, fcos, fsin, qb, kb);
    int nv = BATCH * N_KV * SEQ * HD;
    v_transpose_kernel<<<(nv + 255) / 256, 256, 0, stream>>>(qkvf, vTb);

    // Attention: 1024 blocks; block = one (b,h), 8 waves x 16 query rows
    attn_kernel<<<BATCH * N_HEADS * (SEQ / 128), 256, 0, stream>>>(
        qb, kb, vTb, obuf);

    // Output projection: [4096,2048] x [2048,2048] -> f32 out
    gemm_bf16_kernel<<<dim3(DIM / 128, MTOT / 128), 256, 0, stream>>>(
        obuf, wob, out, MTOT, DIM, DIM);
}
